// TransformerBlock_31207232373314
// MI455X (gfx1250) — matrix operand order repaired and checked by reading, not compiled, not run on hardware
//
#include <hip/hip_runtime.h>
#include <hip/hip_bf16.h>
#include <cstdint>
#include <math.h>

// ---------------- constants ----------------
#define DIM   768
#define HEADS 12
#define HD    64
#define NCTX  2048
#define BATCH 2
#define TOK   (BATCH * NCTX)
#define FFN_DIM (4 * DIM)
#define EPS   1e-6f

typedef __attribute__((ext_vector_type(16))) _Float16 v16h;
typedef __attribute__((ext_vector_type(8)))  float    v8f;
typedef __attribute__((ext_vector_type(4)))  unsigned int u32x4;
typedef __attribute__((ext_vector_type(8)))  int      i32x8;
typedef __attribute__((ext_vector_type(4)))  int      i32x4;

union Frag16 {           // 16 halves == 8 dwords (one WMMA A or B operand)
    v16h     h;
    uint32_t u[8];
};

#if __has_builtin(__builtin_amdgcn_tensor_load_to_lds)
#define HAVE_TDM 1
#else
#define HAVE_TDM 0
#endif

// LDS tile row stride in halves: 64 data halves + 8 pad halves (TDM pad: 4 dwords
// every 32 dwords) -> bank-conflict-reduced fragment reads.
#define TLD 72

#if HAVE_TDM
// Build a D# for a 2D tile [tile_rows x tile_k halves] of a row-major f16 tensor
// with row stride K_elems, and DMA it into LDS at lds_off with padding.
__device__ __forceinline__ void tdm_load_2d(uint32_t lds_off, const _Float16* gptr,
                                            int K_elems, int rows_total,
                                            int tile_k, int tile_rows) {
    const uint64_t ga = (uint64_t)(uintptr_t)gptr;
    u32x4 g0;
    g0[0] = 1u;                                            // count=1, user mode
    g0[1] = lds_off;                                       // LDS byte address
    g0[2] = (uint32_t)ga;                                  // global_addr[31:0]
    g0[3] = (uint32_t)((ga >> 32) & 0x1FFFFFFu) | (2u << 30); // addr[56:32] | type=2
    i32x8 g1;
    const uint32_t d0 = (1u << 16)   // data_size = 2 bytes
                      | (1u << 20)   // pad_enable
                      | (4u << 22)   // pad_interval: 2^(4+1)=32 dwords (=64 halves)
                      | (3u << 25);  // pad_amount: 4 dwords (=8 halves)
    g1[0] = (int)d0;
    g1[1] = (int)(((uint32_t)K_elems & 0xFFFFu) << 16);    // tensor_dim0[15:0]
    g1[2] = (int)(((uint32_t)K_elems >> 16) |
                  (((uint32_t)rows_total & 0xFFFFu) << 16)); // dim0[31:16] | dim1[15:0]
    g1[3] = (int)(((uint32_t)rows_total >> 16) |
                  ((uint32_t)tile_k << 16));               // dim1[31:16] | tile_dim0
    g1[4] = (int)(uint32_t)tile_rows;                      // tile_dim1 | tile_dim2=0
    g1[5] = (int)(uint32_t)K_elems;                        // dim0_stride[31:0]
    g1[6] = 0;                                             // dim0_stride[47:32] | dim1_stride lo
    g1[7] = 0;
    const i32x4 z4 = {0, 0, 0, 0};
    const i32x8 z8 = {0, 0, 0, 0, 0, 0, 0, 0};
    __builtin_amdgcn_tensor_load_to_lds(g0, g1, z4, z4, z8, 0);
}
#endif

// ---------------- weight convert + transpose (fp32 row-major [K,N] -> f16 [N,K]) -----
__global__ void w_to_f16t(const float* __restrict__ W, _Float16* __restrict__ Wt,
                          int K, int N) {
    size_t idx = (size_t)blockIdx.x * 256 + threadIdx.x;
    if (idx >= (size_t)K * N) return;
    int k = (int)(idx / N);
    int n = (int)(idx % N);
    Wt[(size_t)n * K + k] = (_Float16)W[idx];
}

// ---------------- RMSNorm -> f16 ----------------
__global__ __launch_bounds__(256)
void rmsnorm_f16(const float* __restrict__ x, const float* __restrict__ g,
                 _Float16* __restrict__ out) {
    __shared__ float red[256];
    const int tok = blockIdx.x;
    const float* xr = x + (size_t)tok * DIM;
    float s = 0.f;
    for (int i = threadIdx.x; i < DIM; i += 256) { float v = xr[i]; s += v * v; }
    red[threadIdx.x] = s;
    __syncthreads();
    for (int off = 128; off > 0; off >>= 1) {
        if (threadIdx.x < off) red[threadIdx.x] += red[threadIdx.x + off];
        __syncthreads();
    }
    const float scale = rsqrtf(red[0] / (float)DIM + EPS);
    for (int i = threadIdx.x; i < DIM; i += 256)
        out[(size_t)tok * DIM + i] = (_Float16)(xr[i] * scale * g[i]);
}

// ---------------- LDS-tiled WMMA GEMM: C[M,N] = epi(A[M,K] * Bt[N,K]^T) -------------
// Block = 256 threads (8 waves); block tile 128(M) x 64(N); wave tile 32x32 (2x2 frags).
// A/B tiles staged in LDS by the Tensor Data Mover (padded rows, stride TLD halves).
template <typename OutT, int ACT, bool RESID>
__global__ __launch_bounds__(256)
void gemm_wmma(const _Float16* __restrict__ A, const _Float16* __restrict__ Bt,
               const float* __restrict__ bias, const float* __restrict__ resid,
               OutT* __restrict__ C, int M, int N, int K) {
    __shared__ _Float16 sA[128 * TLD];
    __shared__ _Float16 sB[64 * TLD];

    const int lane = threadIdx.x & 31;
    const int wave = threadIdx.x >> 5;
    const int hi   = lane >> 4;
    const int l15  = lane & 15;
    const int mw   = wave & 3;        // 4 m-slots of 32 rows
    const int nw   = wave >> 2;       // 2 n-slots of 32 cols
    const int m0   = blockIdx.y * 128;
    const int n0   = blockIdx.x * 64;

    v8f acc[2][2] = {};

    for (int kb = 0; kb < K; kb += 64) {
        __syncthreads();   // previous iteration done reading LDS
#if HAVE_TDM
        if (wave == 0) {
            tdm_load_2d((uint32_t)(uintptr_t)sA, A + (size_t)m0 * K + kb, K, M, 64, 128);
            tdm_load_2d((uint32_t)(uintptr_t)sB, Bt + (size_t)n0 * K + kb, K, N, 64, 64);
            __builtin_amdgcn_s_wait_tensorcnt(0);
        }
#else
        for (int idx = threadIdx.x; idx < 128 * 32; idx += 256) {
            const int r = idx >> 5, c = idx & 31;        // dword granularity
            *(uint32_t*)(sA + r * TLD + c * 2) =
                *(const uint32_t*)(A + (size_t)(m0 + r) * K + kb + c * 2);
        }
        for (int idx = threadIdx.x; idx < 64 * 32; idx += 256) {
            const int r = idx >> 5, c = idx & 31;
            *(uint32_t*)(sB + r * TLD + c * 2) =
                *(const uint32_t*)(Bt + (size_t)(n0 + r) * K + kb + c * 2);
        }
#endif
        __syncthreads();

#pragma unroll
        for (int k2 = 0; k2 < 64; k2 += 32) {
            Frag16 a0, a1, b0, b1;
#pragma unroll
            for (int j = 0; j < 8; ++j) {
                const int ka = k2 + (hi << 4) + (j << 1);
                a0.u[j] = *(const uint32_t*)(sA + (mw * 32 + l15) * TLD + ka);
                a1.u[j] = *(const uint32_t*)(sA + (mw * 32 + 16 + l15) * TLD + ka);
                const int kk = k2 + (hi << 4) + (j << 1);
                b0.u[j] = *(const uint32_t*)(sB + (nw * 32 + l15) * TLD + kk);
                b1.u[j] = *(const uint32_t*)(sB + (nw * 32 + 16 + l15) * TLD + kk);
            }
            acc[0][0] = __builtin_amdgcn_wmma_f32_16x16x32_f16(false, a0.h, false, b0.h,
                                                               (short)0, acc[0][0], false, false);
            acc[0][1] = __builtin_amdgcn_wmma_f32_16x16x32_f16(false, a0.h, false, b1.h,
                                                               (short)0, acc[0][1], false, false);
            acc[1][0] = __builtin_amdgcn_wmma_f32_16x16x32_f16(false, a1.h, false, b0.h,
                                                               (short)0, acc[1][0], false, false);
            acc[1][1] = __builtin_amdgcn_wmma_f32_16x16x32_f16(false, a1.h, false, b1.h,
                                                               (short)0, acc[1][1], false, false);
        }
    }

#pragma unroll
    for (int ti = 0; ti < 2; ++ti)
#pragma unroll
        for (int tj = 0; tj < 2; ++tj)
#pragma unroll
            for (int i = 0; i < 8; ++i) {
                const int m = m0 + mw * 32 + ti * 16 + (hi << 3) + i;
                const int n = n0 + nw * 32 + tj * 16 + l15;
                float v = acc[ti][tj][i];
                if (bias) v += bias[n];
                if (ACT == 1) v = v / (1.0f + __expf(-v));   // SiLU
                if (RESID) v += resid[(size_t)m * N + n];
                C[(size_t)m * N + n] = (OutT)v;
            }
}

// ---------------- RoPE + head repack ----------------
// q,k -> [b*h, n, 64] f16 ; v -> TRANSPOSED [b*h, 64, n] f16 (contiguous keys)
__global__ __launch_bounds__(256)
void rope_pack(const float* __restrict__ q32, const float* __restrict__ k32,
               const float* __restrict__ v32, const float* __restrict__ cosv,
               const float* __restrict__ sinv, _Float16* __restrict__ q16,
               _Float16* __restrict__ k16, _Float16* __restrict__ vt16) {
    const int idx = blockIdx.x * 256 + threadIdx.x;   // TOK*DIM threads
    const int tok = idx / DIM, c = idx % DIM;
    const int h = c >> 6, d = c & 63;
    const int b = tok / NCTX, n = tok % NCTX;
    const int bh = b * HEADS + h;
    const size_t dst  = ((size_t)bh * NCTX + n) * HD + d;
    const size_t dstv = ((size_t)bh * HD + d) * NCTX + n;   // transposed
    const size_t base = (size_t)tok * DIM + h * HD;
    const float cs = cosv[n * HD + d], sn = sinv[n * HD + d];
    const float qv = q32[base + d];
    const float qr = (d < 32) ? -q32[base + 2 * d + 1] : q32[base + 2 * (d - 32)];
    q16[dst] = (_Float16)(qv * cs + qr * sn);
    const float kv = k32[base + d];
    const float kr = (d < 32) ? -k32[base + 2 * d + 1] : k32[base + 2 * (d - 32)];
    k16[dst] = (_Float16)(kv * cs + kr * sn);
    vt16[dstv] = (_Float16)v32[base + d];
}

// ---------------- attention: one block per (bh, 16-query tile) ----------------
#define SLD (NCTX + 8)
__global__ __launch_bounds__(128)
void attention_kernel(const _Float16* __restrict__ q16, const _Float16* __restrict__ k16,
                      const _Float16* __restrict__ vt16, const float* __restrict__ mask,
                      _Float16* __restrict__ out16) {
    extern __shared__ float smem[];
    float* S      = smem;              // 16 * SLD
    float* red    = S + 16 * SLD;      // 16 * 8
    float* rowmax = red + 16 * 8;      // 16
    float* rowsum = rowmax + 16;       // 16

    const int lane = threadIdx.x & 31;
    const int wave = threadIdx.x >> 5;
    const int hi   = lane >> 4;
    const int l15  = lane & 15;
    const int bh   = blockIdx.y;
    const int q0   = blockIdx.x * 16;
    const size_t hb = (size_t)bh * NCTX * HD;
    const int b = bh / HEADS, h = bh % HEADS;

    // ---- phase 1: S[16, 2048] = (Q Kt) / 8 + mask ----
    const _Float16* qrow = q16 + hb + (size_t)(q0 + l15) * HD;
    for (int kt = wave; kt < NCTX / 16; kt += 4) {
        const _Float16* kcol = k16 + hb + (size_t)(kt * 16 + l15) * HD;
        v8f acc = {};
#pragma unroll
        for (int kb = 0; kb < HD; kb += 32) {
            Frag16 a, bf;
#pragma unroll
            for (int j = 0; j < 8; ++j) {
                const int ka = kb + (hi << 4) + (j << 1);
                a.u[j] = *(const uint32_t*)(qrow + ka);
                const int kk = kb + (hi << 4) + (j << 1);
                bf.u[j] = *(const uint32_t*)(kcol + kk);
            }
            acc = __builtin_amdgcn_wmma_f32_16x16x32_f16(false, a.h, false, bf.h,
                                                         (short)0, acc, false, false);
        }
#pragma unroll
        for (int i = 0; i < 8; ++i) {
            const int r   = (hi << 3) + i;
            const int key = kt * 16 + l15;
            S[r * SLD + key] = acc[i] * 0.125f + mask[(size_t)(q0 + r) * NCTX + key];
        }
    }
    __syncthreads();

    // ---- phase 2: softmax (exp in place, keep rowsum) ----
    {
        const int rrow = threadIdx.x >> 3;
        const int seg  = threadIdx.x & 7;
        float* p = S + rrow * SLD + seg * 256;
        float mx = -3.0e38f;
        for (int j = 0; j < 256; ++j) mx = fmaxf(mx, p[j]);
        red[rrow * 8 + seg] = mx;
        __syncthreads();
        if (seg == 0) {
            float m = red[rrow * 8];
            for (int j = 1; j < 8; ++j) m = fmaxf(m, red[rrow * 8 + j]);
            rowmax[rrow] = m;
        }
        __syncthreads();
        const float m = rowmax[rrow];
        float s = 0.f;
        for (int j = 0; j < 256; ++j) { float e = __expf(p[j] - m); p[j] = e; s += e; }
        red[rrow * 8 + seg] = s;
        __syncthreads();
        if (seg == 0) {
            float t = 0.f;
            for (int j = 0; j < 8; ++j) t += red[rrow * 8 + j];
            rowsum[rrow] = t;
        }
        __syncthreads();
    }

    // ---- phase 3: O[16,64] = P x V (V transposed -> contiguous dword B loads) ----
    {
        const int d0 = wave * 16;
        const _Float16* vcol = vt16 + hb + (size_t)(d0 + l15) * NCTX;
        v8f acc = {};
        for (int kb = 0; kb < NCTX; kb += 32) {
            Frag16 a, bf;
#pragma unroll
            for (int j = 0; j < 8; ++j) {
                const int ka = kb + (hi << 4) + (j << 1);
                a.h[2 * j]     = (_Float16)S[l15 * SLD + ka];
                a.h[2 * j + 1] = (_Float16)S[l15 * SLD + ka + 1];
                const int kk = kb + (hi << 4) + (j << 1);
                bf.u[j] = *(const uint32_t*)(vcol + kk);
            }
            acc = __builtin_amdgcn_wmma_f32_16x16x32_f16(false, a.h, false, bf.h,
                                                         (short)0, acc, false, false);
        }
#pragma unroll
        for (int i = 0; i < 8; ++i) {
            const int r    = (hi << 3) + i;
            const int tokr = b * NCTX + q0 + r;
            out16[(size_t)tokr * DIM + h * HD + d0 + l15] =
                (_Float16)(acc[i] / rowsum[r]);
        }
    }
}

// ---------------- host-side orchestration ----------------
static inline char* alignp(char* p) {
    return (char*)(((uintptr_t)p + 255) & ~(uintptr_t)255);
}

extern "C" void kernel_launch(void* const* d_in, const int* in_sizes, int n_in,
                              void* d_out, int out_size, void* d_ws, size_t ws_size,
                              hipStream_t stream) {
    const float* x      = (const float*)d_in[0];
    const float* cosv   = (const float*)d_in[1];
    const float* sinv   = (const float*)d_in[2];
    const float* mask   = (const float*)d_in[3];
    const float* wq     = (const float*)d_in[4];
    const float* wk     = (const float*)d_in[5];
    const float* wv     = (const float*)d_in[6];
    const float* wo     = (const float*)d_in[7];
    const float* bo     = (const float*)d_in[8];
    const float* g_attn = (const float*)d_in[9];
    const float* g_ffn  = (const float*)d_in[10];
    const float* w1     = (const float*)d_in[11];
    const float* b1     = (const float*)d_in[12];
    const float* w2     = (const float*)d_in[13];
    const float* b2     = (const float*)d_in[14];
    float* out = (float*)d_out;

    // ---- carve workspace ----
    char* p = (char*)d_ws;
    auto carve = [&](size_t bytes) { char* r = alignp(p); p = r + bytes; return r; };
    _Float16* wq16  = (_Float16*)carve((size_t)DIM * DIM * 2);
    _Float16* wk16  = (_Float16*)carve((size_t)DIM * DIM * 2);
    _Float16* wv16  = (_Float16*)carve((size_t)DIM * DIM * 2);
    _Float16* wo16  = (_Float16*)carve((size_t)DIM * DIM * 2);
    _Float16* w116  = (_Float16*)carve((size_t)DIM * FFN_DIM * 2);
    _Float16* w216  = (_Float16*)carve((size_t)FFN_DIM * DIM * 2);
    _Float16* h16   = (_Float16*)carve((size_t)TOK * DIM * 2);
    float*    q32   = (float*)   carve((size_t)TOK * DIM * 4);
    float*    k32   = (float*)   carve((size_t)TOK * DIM * 4);
    float*    v32   = (float*)   carve((size_t)TOK * DIM * 4);
    _Float16* q16   = (_Float16*)carve((size_t)TOK * DIM * 2);
    _Float16* k16   = (_Float16*)carve((size_t)TOK * DIM * 2);
    _Float16* vt16  = (_Float16*)carve((size_t)TOK * DIM * 2);
    _Float16* att16 = (_Float16*)carve((size_t)TOK * DIM * 2);
    float*    x2    = (float*)   carve((size_t)TOK * DIM * 4);
    _Float16* h216  = (_Float16*)carve((size_t)TOK * DIM * 2);
    _Float16* g116  = (_Float16*)carve((size_t)TOK * FFN_DIM * 2);

    // ---- 1: weights -> f16 transposed ----
    {
        const int gsq = (DIM * DIM + 255) / 256;
        const int gw1 = (DIM * FFN_DIM + 255) / 256;
        w_to_f16t<<<gsq, 256, 0, stream>>>(wq, wq16, DIM, DIM);
        w_to_f16t<<<gsq, 256, 0, stream>>>(wk, wk16, DIM, DIM);
        w_to_f16t<<<gsq, 256, 0, stream>>>(wv, wv16, DIM, DIM);
        w_to_f16t<<<gsq, 256, 0, stream>>>(wo, wo16, DIM, DIM);
        w_to_f16t<<<gw1, 256, 0, stream>>>(w1, w116, DIM, FFN_DIM);
        w_to_f16t<<<gw1, 256, 0, stream>>>(w2, w216, FFN_DIM, DIM);
    }

    // ---- 2: rmsnorm(x, g_attn) -> h16 ----
    rmsnorm_f16<<<TOK, 256, 0, stream>>>(x, g_attn, h16);

    // ---- 3: Q/K/V = h @ w{q,k,v} ----
    {
        dim3 grid(DIM / 64, TOK / 128), blk(256);
        gemm_wmma<float, 0, false><<<grid, blk, 0, stream>>>(h16, wq16, nullptr, nullptr, q32, TOK, DIM, DIM);
        gemm_wmma<float, 0, false><<<grid, blk, 0, stream>>>(h16, wk16, nullptr, nullptr, k32, TOK, DIM, DIM);
        gemm_wmma<float, 0, false><<<grid, blk, 0, stream>>>(h16, wv16, nullptr, nullptr, v32, TOK, DIM, DIM);
    }

    // ---- 4: RoPE + repack (q,k row-major; v transposed per head) ----
    rope_pack<<<(TOK * DIM) / 256, 256, 0, stream>>>(q32, k32, v32, cosv, sinv, q16, k16, vt16);

    // ---- 5: attention ----
    {
        dim3 grid(NCTX / 16, BATCH * HEADS), blk(128);
        const size_t smem = (size_t)(16 * SLD + 16 * 8 + 16 + 16) * sizeof(float);
        attention_kernel<<<grid, blk, smem, stream>>>(q16, k16, vt16, mask, att16);
    }

    // ---- 6: x2 = x + att @ wo + bo ----
    {
        dim3 grid(DIM / 64, TOK / 128), blk(256);
        gemm_wmma<float, 0, true><<<grid, blk, 0, stream>>>(att16, wo16, bo, x, x2, TOK, DIM, DIM);
    }

    // ---- 7: rmsnorm(x2, g_ffn) -> h216 ----
    rmsnorm_f16<<<TOK, 256, 0, stream>>>(x2, g_ffn, h216);

    // ---- 8: g1 = silu(h2 @ w1 + b1) ----
    {
        dim3 grid(FFN_DIM / 64, TOK / 128), blk(256);
        gemm_wmma<_Float16, 1, false><<<grid, blk, 0, stream>>>(h216, w116, b1, nullptr, g116, TOK, FFN_DIM, DIM);
    }

    // ---- 9: out = x2 + g1 @ w2 + b2 ----
    {
        dim3 grid(DIM / 64, TOK / 128), blk(256);
        gemm_wmma<float, 0, true><<<grid, blk, 0, stream>>>(g116, w216, b2, x2, out, TOK, DIM, FFN_DIM);
    }

    (void)in_sizes; (void)n_in; (void)out_size; (void)ws_size;
}
